// MultiHeadAttention_17145509446314
// MI455X (gfx1250) — compile-verified, matrix operand order checked by reading, not run on hardware
//
#include <hip/hip_runtime.h>
#include <hip/hip_bf16.h>

// ---------------------------------------------------------------------------
// MultiHeadAttention for MI455X (gfx1250): f16 WMMA, f32 accumulate,
// flash-attention fused softmax, causal mask analytic, TDM (tensor_load_to_lds)
// staging of K/V tiles in the attention kernel, ping-pong pipelined GEMMs
// (no register rotation -> no v_mov storm in the hot loop).
// Shapes: B=2, S=2048, D_MODEL=1024, H=16, dk=dv=64.
// ---------------------------------------------------------------------------

typedef __attribute__((ext_vector_type(16))) _Float16     v16h;
typedef __attribute__((ext_vector_type(8)))  _Float16     v8h;
typedef __attribute__((ext_vector_type(8)))  float        v8f;
typedef __attribute__((ext_vector_type(4)))  unsigned int v4u;
typedef __attribute__((ext_vector_type(8)))  int          v8i;
typedef __attribute__((ext_vector_type(4)))  int          v4i;

static constexpr int S_LEN   = 2048;
static constexpr int D_MODEL = 1024;
static constexpr int NHEADS  = 16;
static constexpr int DHEAD   = 64;
static constexpr int MROWS   = 2 * S_LEN;       // 4096 total rows (B*S)

#if __has_builtin(__builtin_amdgcn_tensor_load_to_lds)
#define HAVE_TDM 1
#else
#define HAVE_TDM 0
#endif

// Load one 16x32 f16 A/B WMMA fragment whose per-lane K-run is contiguous in
// memory: halves 0..7 <- K=base+0..7, halves 8..15 <- K=base+16..23 (caller
// folds the +8 for lanes 16..31 into p).  Two 16B loads -> *_load_b128.
__device__ inline v16h load_frag(const _Float16* p) {
  union { v16h v; v8h h[2]; } u;
  u.h[0] = *(const v8h*)(p);
  u.h[1] = *(const v8h*)(p + 16);
  return u.v;
}

__device__ inline v8f wmma_f16(v16h a, v16h b, v8f c) {
  return __builtin_amdgcn_wmma_f32_16x16x32_f16(false, a, false, b,
                                                (short)0, c, false, false);
}

#if HAVE_TDM
// Issue one TDM 2D tile load: global (row-major, stride0 elements between
// rows, 2-byte elements) -> LDS, with LDS row padding of 4 DWORDs (8 halves)
// every 32 DWORDs (64 halves) so staged rows land on a 72-half stride.
// D# field packing per CDNA5 ISA ch.7/8.
__device__ inline void tdm_load_2d(unsigned lds_addr, const void* gaddr,
                                   unsigned tile_d0, unsigned tile_d1,
                                   unsigned tensor_d0, unsigned tensor_d1,
                                   unsigned long long stride0) {
  unsigned long long ga = (unsigned long long)gaddr;
  v4u g0;
  g0[0] = 1u;                                        // count=1 (valid user D#)
  g0[1] = lds_addr;                                  // lds_addr[31:0]
  g0[2] = (unsigned)(ga & 0xffffffffu);              // global_addr[31:0]
  g0[3] = (unsigned)((ga >> 32) & 0x01ffffffu)       // global_addr[56:32]
        | 0x80000000u;                               // type=2 ("image")
  unsigned w[8] = {};
  w[0]  = (1u << 16)                                 // data_size=1 (2 bytes)
        | (1u << 20)                                 // pad_enable
        | (4u << 22)                                 // pad_interval: 32 DWORDs
        | (3u << 25);                                // pad_amount: 4 DWORDs
  w[1] |= (tensor_d0 & 0xffffu) << 16;               // tensor_dim0[15:0]
  w[2] |= (tensor_d0 >> 16) & 0xffffu;               // tensor_dim0[31:16]
  w[2] |= (tensor_d1 & 0xffffu) << 16;               // tensor_dim1[15:0]
  w[3] |= (tensor_d1 >> 16) & 0xffffu;               // tensor_dim1[31:16]
  w[3] |= (tile_d0 & 0xffffu) << 16;                 // tile_dim0
  w[4] |= (tile_d1 & 0xffffu);                       // tile_dim1
  w[5]  = (unsigned)(stride0 & 0xffffffffu);         // tensor_dim0_stride lo
  w[6]  = (unsigned)((stride0 >> 32) & 0xffffu);     // tensor_dim0_stride hi
  v8i g1 = { (int)w[0], (int)w[1], (int)w[2], (int)w[3],
             (int)w[4], (int)w[5], (int)w[6], (int)w[7] };
  v4i z4 = {0, 0, 0, 0};
#if __clang_major__ >= 23
  v8i z8 = {0, 0, 0, 0, 0, 0, 0, 0};
  __builtin_amdgcn_tensor_load_to_lds(g0, g1, z4, z4, z8, 0);
#else
  __builtin_amdgcn_tensor_load_to_lds(g0, g1, z4, z4, 0);
#endif
}
#endif  // HAVE_TDM

__device__ inline void wait_tensorcnt0() {
#if __has_builtin(__builtin_amdgcn_s_wait_tensorcnt)
  __builtin_amdgcn_s_wait_tensorcnt(0);
#else
  asm volatile("s_wait_tensorcnt 0x0" ::: "memory");
#endif
}

// ---------------------------------------------------------------------------
// Stage 1a: flat f32 -> f16 convert (activations)
// ---------------------------------------------------------------------------
__global__ void __launch_bounds__(256)
cvt_f32_to_f16(const float* __restrict__ in, _Float16* __restrict__ out, int n) {
  int i = (blockIdx.x * 256 + threadIdx.x) * 4;
  if (i + 3 < n) {
    float4 f = *(const float4*)(in + i);
    out[i + 0] = (_Float16)f.x;
    out[i + 1] = (_Float16)f.y;
    out[i + 2] = (_Float16)f.z;
    out[i + 3] = (_Float16)f.w;
  }
}

// ---------------------------------------------------------------------------
// Stage 1b: 1024x1024 f32 -> f16 transpose (weights), LDS-tiled.
// ---------------------------------------------------------------------------
__global__ void __launch_bounds__(256)
transpose_cvt(const float* __restrict__ in, _Float16* __restrict__ out) {
  __shared__ float tile[32][33];
  int x = blockIdx.x * 32 + threadIdx.x;
  int y = blockIdx.y * 32 + threadIdx.y;
#pragma unroll
  for (int i = 0; i < 32; i += 8)
    tile[threadIdx.y + i][threadIdx.x] = in[(size_t)(y + i) * D_MODEL + x];
  __syncthreads();
  int ox = blockIdx.y * 32 + threadIdx.x;
  int oy = blockIdx.x * 32 + threadIdx.y;
#pragma unroll
  for (int i = 0; i < 32; i += 8)
    out[(size_t)(oy + i) * D_MODEL + ox] = (_Float16)tile[threadIdx.x][threadIdx.y + i];
}

// ---------------------------------------------------------------------------
// Stage 2/5: pipelined GEMM  C[4096x1024] = A[4096x1024] * W (W transposed).
// 8 waves/block, wave tile 16x64.  Ping-pong double buffering via 2x unroll:
// two named fragment sets alternate, so the back-edge is a reload (no copies).
// mode 0: f16 head-major [BH][S][64]; mode 1: f16 [BH][64][S]; mode 2: f32.
// ---------------------------------------------------------------------------
__global__ void __launch_bounds__(256)
gemm_wmma(const _Float16* __restrict__ A, const _Float16* __restrict__ Wt,
          _Float16* __restrict__ outH, float* __restrict__ outF, int mode) {
  const int lane = threadIdx.x & 31;
  const int wave = threadIdx.x >> 5;
  const int hi   = lane >> 4;
  const int ln   = lane & 15;
  const int rowBase = blockIdx.x * 128 + wave * 16;
  const int colBase = blockIdx.y * 64;

  v8f acc[4] = {};

  const _Float16* arow = A + (size_t)(rowBase + ln) * D_MODEL + hi * 8;
  const _Float16* brow[4];
#pragma unroll
  for (int nt = 0; nt < 4; ++nt)
    brow[nt] = Wt + (size_t)(colBase + nt * 16 + ln) * D_MODEL + hi * 8;

  // prologue: buffer 0 holds kb = 0
  v16h af0 = load_frag(arow);
  v16h bf0[4];
#pragma unroll
  for (int nt = 0; nt < 4; ++nt) bf0[nt] = load_frag(brow[nt]);

  int kb = 0;
  for (; kb < D_MODEL - 64; kb += 64) {
    // buffer 1 <- kb+32 (in flight under buf0's WMMAs)
    v16h af1 = load_frag(arow + kb + 32);
    v16h bf1[4];
#pragma unroll
    for (int nt = 0; nt < 4; ++nt) bf1[nt] = load_frag(brow[nt] + kb + 32);
#pragma unroll
    for (int nt = 0; nt < 4; ++nt) acc[nt] = wmma_f16(af0, bf0[nt], acc[nt]);
    // buffer 0 <- kb+64 (in flight under buf1's WMMAs); back-edge phi, no movs
    af0 = load_frag(arow + kb + 64);
#pragma unroll
    for (int nt = 0; nt < 4; ++nt) bf0[nt] = load_frag(brow[nt] + kb + 64);
#pragma unroll
    for (int nt = 0; nt < 4; ++nt) acc[nt] = wmma_f16(af1, bf1[nt], acc[nt]);
  }
  // epilogue: kb = D_MODEL-64; buf0 holds kb, load kb+32, drain both
  {
    v16h af1 = load_frag(arow + kb + 32);
    v16h bf1[4];
#pragma unroll
    for (int nt = 0; nt < 4; ++nt) bf1[nt] = load_frag(brow[nt] + kb + 32);
#pragma unroll
    for (int nt = 0; nt < 4; ++nt) acc[nt] = wmma_f16(af0, bf0[nt], acc[nt]);
#pragma unroll
    for (int nt = 0; nt < 4; ++nt) acc[nt] = wmma_f16(af1, bf1[nt], acc[nt]);
  }

#pragma unroll
  for (int nt = 0; nt < 4; ++nt) {
#pragma unroll
    for (int r = 0; r < 8; ++r) {
      int m   = r + hi * 8;
      int g   = rowBase + m;
      int col = colBase + nt * 16 + ln;
      float v = acc[nt][r];
      if (mode == 2) {
        outF[(size_t)g * D_MODEL + col] = v;
      } else {
        int b = g >> 11, s = g & (S_LEN - 1);
        int h = col >> 6, d = col & (DHEAD - 1);
        size_t bh = (size_t)(b * NHEADS + h);
        if (mode == 0)
          outH[(bh * S_LEN + s) * DHEAD + d] = (_Float16)v;   // [BH][S][64]
        else
          outH[(bh * DHEAD + d) * S_LEN + s] = (_Float16)v;   // [BH][64][S]
      }
    }
  }
}

// ---------------------------------------------------------------------------
// Stage 3: fused causal flash attention.  One block = (64 q rows, batch*head).
// 4 waves, each owns a 16-row Q stripe in registers.  Per key block wave 0
// TDM-DMAs the 64x64 K and V tiles into padded LDS (s_wait_tensorcnt +
// barrier), all waves read fragments via ds_load_b128.  Online softmax in
// base-2 domain; P converted C-layout -> A-layout through a per-wave LDS tile.
// ---------------------------------------------------------------------------
__global__ void __launch_bounds__(128)
attn_flash(const _Float16* __restrict__ Qh, const _Float16* __restrict__ Kh,
           const _Float16* __restrict__ Vt, _Float16* __restrict__ Oh) {
  __shared__ alignas(16) _Float16 kbuf[64][72];     // keys  x d   (+8 pad)
  __shared__ alignas(16) _Float16 vbuf[64][72];     // d     x keys(+8 pad)
  __shared__ alignas(16) _Float16 pbuf[4][16][72];  // per-wave P tile

  const int lane  = threadIdx.x & 31;
  const int wave  = threadIdx.x >> 5;
  const int hi    = lane >> 4;
  const int ln    = lane & 15;
  const int qTile = blockIdx.x;          // 0..31
  const int bh    = blockIdx.y;          // 0..31
  const int b     = bh >> 4, h = bh & 15;
  const int qRow0 = qTile * 64 + wave * 16;

  const _Float16* Qb = Qh + ((size_t)bh * S_LEN + qRow0 + ln) * DHEAD + hi * 8;
  v16h aq0 = load_frag(Qb);
  v16h aq1 = load_frag(Qb + 32);

  v8f accO[4] = {};
  float rowm[8], rowl[8];
#pragma unroll
  for (int r = 0; r < 8; ++r) { rowm[r] = -3.0e38f; rowl[r] = 0.0f; }

  const float sc = 0.125f * 1.44269504088896f;   // 1/sqrt(64) * log2(e)

  for (int j = 0; j <= qTile; ++j) {
    const int keyBase = j * 64;
    const _Float16* Kg = Kh + ((size_t)bh * S_LEN + keyBase) * DHEAD;
    const _Float16* Vg = Vt + (size_t)bh * DHEAD * S_LEN + keyBase;

#if HAVE_TDM
    if (wave == 0) {
      tdm_load_2d((unsigned)(size_t)&kbuf[0][0], Kg, 64, 64, 64, 64, DHEAD);
      tdm_load_2d((unsigned)(size_t)&vbuf[0][0], Vg, 64, 64, 64, 64, S_LEN);
      wait_tensorcnt0();
    }
#else
    for (int t = threadIdx.x; t < 512; t += 128) {   // 512 x 16B per buffer
      int d = t >> 3, part = t & 7;
      *(v8h*)&kbuf[d][part * 8] = *(const v8h*)(Kg + (size_t)d * DHEAD + part * 8);
      *(v8h*)&vbuf[d][part * 8] = *(const v8h*)(Vg + (size_t)d * S_LEN + part * 8);
    }
#endif
    __syncthreads();

    // ---- S = Q K^T (K fragments from LDS) ----
    v8f accS[4];
#pragma unroll
    for (int nt = 0; nt < 4; ++nt) {
      v16h bk0 = load_frag(&kbuf[nt * 16 + ln][hi * 8]);
      v16h bk1 = load_frag(&kbuf[nt * 16 + ln][32 + hi * 8]);
      v8f c = {};
      c = wmma_f16(aq0, bk0, c);
      c = wmma_f16(aq1, bk1, c);
      accS[nt] = c;
    }

    // ---- scale (+ causal mask on diagonal block) ----
    const bool diag = (j == qTile);
#pragma unroll
    for (int nt = 0; nt < 4; ++nt) {
#pragma unroll
      for (int r = 0; r < 8; ++r) {
        float s = accS[nt][r] * sc;
        if (diag) {
          int qi = qRow0 + r + hi * 8;
          int ki = keyBase + nt * 16 + ln;
          if (ki > qi) s = -3.0e38f;
        }
        accS[nt][r] = s;
      }
    }

    // ---- online softmax ----
    float nm[8];
#pragma unroll
    for (int r = 0; r < 8; ++r) {
      float m4 = fmaxf(fmaxf(accS[0][r], accS[1][r]),
                       fmaxf(accS[2][r], accS[3][r]));
      m4 = fmaxf(m4, __shfl_xor(m4, 1));
      m4 = fmaxf(m4, __shfl_xor(m4, 2));
      m4 = fmaxf(m4, __shfl_xor(m4, 4));
      m4 = fmaxf(m4, __shfl_xor(m4, 8));   // xor<16 stays inside the N-group
      nm[r] = fmaxf(rowm[r], m4);
    }
#pragma unroll
    for (int r = 0; r < 8; ++r) {
      float alpha = __builtin_amdgcn_exp2f(rowm[r] - nm[r]);
      rowm[r] = nm[r];
      float rs = 0.0f;
#pragma unroll
      for (int nt = 0; nt < 4; ++nt) {
        float p = __builtin_amdgcn_exp2f(accS[nt][r] - nm[r]);
        accS[nt][r] = p;
        rs += p;
        accO[nt][r] *= alpha;
      }
      rs += __shfl_xor(rs, 1);
      rs += __shfl_xor(rs, 2);
      rs += __shfl_xor(rs, 4);
      rs += __shfl_xor(rs, 8);
      rowl[r] = rowl[r] * alpha + rs;
    }

    // ---- P: C layout -> A-fragment layout via per-wave LDS tile ----
#pragma unroll
    for (int nt = 0; nt < 4; ++nt)
#pragma unroll
      for (int r = 0; r < 8; ++r)
        pbuf[wave][r + hi * 8][nt * 16 + ln] = (_Float16)accS[nt][r];
    __syncthreads();
    v16h ap0 = load_frag(&pbuf[wave][ln][hi * 8]);
    v16h ap1 = load_frag(&pbuf[wave][ln][32 + hi * 8]);

    // ---- O += P V (V fragments from LDS) ----
#pragma unroll
    for (int nt = 0; nt < 4; ++nt) {
      v16h bv0 = load_frag(&vbuf[nt * 16 + ln][hi * 8]);
      v16h bv1 = load_frag(&vbuf[nt * 16 + ln][32 + hi * 8]);
      accO[nt] = wmma_f16(ap0, bv0, accO[nt]);
      accO[nt] = wmma_f16(ap1, bv1, accO[nt]);
    }
    __syncthreads();   // all reads of kbuf/vbuf done before next TDM overwrite
  }

  // ---- epilogue: normalize, store merged-head f16 [B][S][H*64] ----
#pragma unroll
  for (int nt = 0; nt < 4; ++nt) {
#pragma unroll
    for (int r = 0; r < 8; ++r) {
      int m = r + hi * 8;
      int s = qRow0 + m;
      int d = nt * 16 + ln;
      float v = accO[nt][r] / rowl[r];
      Oh[((size_t)b * S_LEN + s) * D_MODEL + h * DHEAD + d] = (_Float16)v;
    }
  }
}

// ---------------------------------------------------------------------------
// Host-side orchestration (graph-capture safe)
// ---------------------------------------------------------------------------
extern "C" void kernel_launch(void* const* d_in, const int* in_sizes, int n_in,
                              void* d_out, int out_size, void* d_ws, size_t ws_size,
                              hipStream_t stream) {
  const float* q  = (const float*)d_in[0];
  const float* k  = (const float*)d_in[1];
  const float* v  = (const float*)d_in[2];
  // d_in[3] = boolean causal mask: applied analytically inside attn_flash
  const float* wq = (const float*)d_in[4];
  const float* wk = (const float*)d_in[5];
  const float* wv = (const float*)d_in[6];
  const float* wo = (const float*)d_in[7];
  float* out = (float*)d_out;

  _Float16* ws = (_Float16*)d_ws;
  const size_t M1 = 1024u * 1024u;
  _Float16* Xq  = ws;                    // 4M halves each
  _Float16* Xk  = ws + 4 * M1;
  _Float16* Xv  = ws + 8 * M1;
  _Float16* WqT = ws + 12 * M1;          // 1M each
  _Float16* WkT = ws + 13 * M1;
  _Float16* WvT = ws + 14 * M1;
  _Float16* WoT = ws + 15 * M1;
  _Float16* Qh  = ws + 16 * M1;          // [BH][S][64]
  _Float16* Kh  = ws + 20 * M1;          // [BH][S][64]
  _Float16* Vt  = ws + 24 * M1;          // [BH][64][S]
  _Float16* Oh  = ws + 28 * M1;          // [B][S][1024]

  const int n = MROWS * D_MODEL;
  cvt_f32_to_f16<<<n / 1024, 256, 0, stream>>>(q, Xq, n);
  cvt_f32_to_f16<<<n / 1024, 256, 0, stream>>>(k, Xk, n);
  cvt_f32_to_f16<<<n / 1024, 256, 0, stream>>>(v, Xv, n);

  dim3 tb(32, 8), tg(32, 32);
  transpose_cvt<<<tg, tb, 0, stream>>>(wq, WqT);
  transpose_cvt<<<tg, tb, 0, stream>>>(wk, WkT);
  transpose_cvt<<<tg, tb, 0, stream>>>(wv, WvT);
  transpose_cvt<<<tg, tb, 0, stream>>>(wo, WoT);

  dim3 gg(MROWS / 128, D_MODEL / 64);    // (32, 16)
  gemm_wmma<<<gg, 256, 0, stream>>>(Xq, WqT, Qh, nullptr, 0);
  gemm_wmma<<<gg, 256, 0, stream>>>(Xk, WkT, Kh, nullptr, 0);
  gemm_wmma<<<gg, 256, 0, stream>>>(Xv, WvT, Vt, nullptr, 1);

  attn_flash<<<dim3(S_LEN / 64, 2 * NHEADS), 128, 0, stream>>>(Qh, Kh, Vt, Oh);

  gemm_wmma<<<gg, 256, 0, stream>>>(Oh, WoT, nullptr, out, 2);
}
